// GQA_9715216023893
// MI455X (gfx1250) — matrix operand order repaired and checked by reading, not compiled, not run on hardware
//
#include <hip/hip_runtime.h>

typedef _Float16 h16;
typedef __attribute__((ext_vector_type(16))) _Float16 v16h;
typedef __attribute__((ext_vector_type(8)))  _Float16 v8h;
typedef __attribute__((ext_vector_type(8)))  float    v8f;

#define BB    2
#define NSEQ  2048
#define EDIM  1024
#define HQn   16
#define HKn   4
#define DH    64
#define KVE   256
#define MROWS (BB * NSEQ)          /* 4096 tokens */
#define NWORDS 64                  /* NSEQ/32 mask words per row */
#define NEGMIN (-3.4028234663852886e38f)

// ---------------------------------------------------------------------------
// WMMA fragment helpers (wave32, v_wmma_f32_16x16x32_f16)
// A-frag: lane l -> row (l&15), K = {kb..kb+7, 16+kb..16+kb+7}, kb=(l>>4)*8
// B-frag: lane l -> col (l&15), K = {kb..kb+7, 16+kb..16+kb+7}, kb=(l>>4)*8
// C-frag: vgpr r -> row r + 8*(l>>4), col (l&15)
// ---------------------------------------------------------------------------
__device__ __forceinline__ v16h load_a_frag(const h16* base, int ld, int lane) {
  const int m  = lane & 15;
  const int kb = (lane >> 4) << 3;
  const h16* p = base + (size_t)m * ld + kb;
  v8h lo = *(const v8h*)(p);
  v8h hi = *(const v8h*)(p + 16);
  v16h a;
#pragma unroll
  for (int i = 0; i < 8; ++i) { a[i] = lo[i]; a[8 + i] = hi[i]; }
  return a;
}

__device__ __forceinline__ v16h load_b_frag(const h16* base, int ld, int lane) {
  const int n  = lane & 15;
  const int kb = (lane >> 4) << 3;
  const h16* p = base + (size_t)n * ld + kb;
  v8h lo = *(const v8h*)(p);
  v8h hi = *(const v8h*)(p + 16);
  v16h b;
#pragma unroll
  for (int i = 0; i < 8; ++i) { b[i] = lo[i]; b[8 + i] = hi[i]; }
  return b;
}

__device__ __forceinline__ v8f wmma_f16(v16h a, v16h b, v8f c) {
  return __builtin_amdgcn_wmma_f32_16x16x32_f16(false, a, false, b,
                                                (short)0, c, false, false);
}

// XOR lane exchange via ds_swizzle_b32 (group-of-32: xor<<10 | and 0x1F).
// Masks <= 8 stay inside each 16-lane half (the C-frag row-reduction domain).
template <int XM>
__device__ __forceinline__ float swz_xor(float v) {
  return __int_as_float(
      __builtin_amdgcn_ds_swizzle(__float_as_int(v), (XM << 10) | 0x1F));
}

__device__ __forceinline__ float half_reduce_max(float t) {
  t = fmaxf(t, swz_xor<1>(t));
  t = fmaxf(t, swz_xor<2>(t));
  t = fmaxf(t, swz_xor<4>(t));
  t = fmaxf(t, swz_xor<8>(t));
  return t;
}

__device__ __forceinline__ float half_reduce_sum(float t) {
  t += swz_xor<1>(t);
  t += swz_xor<2>(t);
  t += swz_xor<4>(t);
  t += swz_xor<8>(t);
  return t;
}

// ---------------------------------------------------------------------------
// fp32 -> fp16 conversion
// ---------------------------------------------------------------------------
__global__ __launch_bounds__(256) void cvt_f32_f16(const float* __restrict__ s,
                                                   h16* __restrict__ d, int n) {
  int i = blockIdx.x * blockDim.x + threadIdx.x;
  if (i < n) d[i] = (h16)s[i];
}

// ---------------------------------------------------------------------------
// Pack (causal & adj) into bitmask words: bit i of word (b,n,w) covers
// s = w*32+i. Removes all causal compares / adj loads from the hot loop.
// ---------------------------------------------------------------------------
__global__ __launch_bounds__(256) void pack_mask(const int* __restrict__ adj,
                                                 unsigned* __restrict__ mw) {
  const int idx = blockIdx.x * 256 + threadIdx.x;   // B*N*NWORDS = 262144
  const int w = idx & (NWORDS - 1);
  const int n = (idx >> 6) & (NSEQ - 1);
  const int b = idx >> 17;
  const int* row = adj + ((size_t)b * NSEQ + n) * NSEQ + w * 32;
  unsigned bits = 0;
#pragma unroll
  for (int i = 0; i < 32; ++i) bits |= (row[i] != 0 ? 1u : 0u) << i;
  const int s0 = w * 32;
  unsigned causal;
  if (n >= s0 + 31)      causal = 0xffffffffu;
  else if (n < s0)       causal = 0u;
  else                   causal = (2u << (n - s0)) - 1u;
  mw[idx] = bits & causal;
}

// ---------------------------------------------------------------------------
// GEMM: out = A(MxK,f16) @ W(NoutxK,f16)^T + bias   (one wave per 16x64 strip)
// MODE 0: f16 out, row-major, scaled      (Q / K projections)
// MODE 1: f16 out, transposed per token-batch: out[(b*KVE+n)*NSEQ + s]  (V)
// MODE 2: f32 out, row-major              (final output projection)
// ---------------------------------------------------------------------------
template <int MODE>
__global__ __launch_bounds__(256) void gemm_wmma(
    const h16* __restrict__ A, const h16* __restrict__ W,
    const float* __restrict__ bias, void* __restrict__ outp,
    int M, int K, int Nout, float scale) {
  const int lane = threadIdx.x & 31;
  const int wid  = blockIdx.x * (blockDim.x >> 5) + (threadIdx.x >> 5);
  const int nstrips = Nout >> 6;
  const int mt = wid / nstrips;
  const int ns = wid % nstrips;
  if (mt * 16 >= M) return;
  const int m0 = mt * 16, n0 = ns * 64;

  v8f acc[4] = {};
  for (int k0 = 0; k0 < K; k0 += 32) {
    v16h af = load_a_frag(A + (size_t)m0 * K + k0, K, lane);
#pragma unroll
    for (int t = 0; t < 4; ++t) {
      v16h bf = load_b_frag(W + (size_t)(n0 + t * 16) * K + k0, K, lane);
      acc[t] = wmma_f16(af, bf, acc[t]);
    }
  }

  const int nl = lane & 15;
  const int mb = (lane >> 4) << 3;
#pragma unroll
  for (int t = 0; t < 4; ++t) {
#pragma unroll
    for (int r = 0; r < 8; ++r) {
      const int n = n0 + t * 16 + nl;
      const int m = m0 + mb + r;
      const float v = acc[t][r] + bias[n];
      if (MODE == 2) {
        ((float*)outp)[(size_t)m * Nout + n] = v;
      } else if (MODE == 1) {
        const int b = m / NSEQ, s = m % NSEQ;  // v^T: [b][feature][seq]
        ((h16*)outp)[((size_t)(b * KVE + n)) * NSEQ + s] = (h16)v;
      } else {
        ((h16*)outp)[(size_t)m * Nout + n] = (h16)(v * scale);
      }
    }
  }
}

// ---------------------------------------------------------------------------
// Flash-style GQA attention. Block = 4 waves, all on the SAME (b, hq) with
// consecutive 16-row query tiles, so K/V tiles are staged in LDS once per
// block and shared (4x cut in global traffic). Sweeps the full S axis to
// match reference masking semantics (finite neg -> uniform rows).
// ---------------------------------------------------------------------------
__global__ __launch_bounds__(128) void attn_wmma(
    const h16* __restrict__ qh,       // [B*N, E]  (pre-scaled by 1/sqrt(D))
    const h16* __restrict__ kh,       // [B*N, KVE]
    const h16* __restrict__ vth,      // [(b*KVE+feat)*N + s]
    const unsigned* __restrict__ mw,  // [B*N, NWORDS] packed causal&adj
    float* __restrict__ attn_out)     // [B*N, E]
{
  const int tid  = threadIdx.x;
  const int lane = tid & 31;
  const int w    = tid >> 5;
  const int hq   = (blockIdx.x >> 5) & (HQn - 1);   // same for all 4 waves
  const int b    = blockIdx.x >> 9;
  const int h    = hq >> 2;                          // kv head = hq / G
  const int n0   = ((blockIdx.x * 4 + w) & (NSEQ / 16 - 1)) * 16;

  __shared__ __align__(16) h16 ktile[32 * 64];       // [s][d]  ld=64
  __shared__ __align__(16) h16 vtile[64 * 32];       // [d][s]  ld=32
  __shared__ __align__(16) h16 lds_p[4][16 * 32];    // per-wave P tiles
  h16* pt = lds_p[w];

  const h16* qbase = qh + ((size_t)(b * NSEQ + n0)) * EDIM + hq * DH;
  const v16h qa0 = load_a_frag(qbase, EDIM, lane);
  const v16h qa1 = load_a_frag(qbase + 32, EDIM, lane);

  float m_run[8], l_run[8];
#pragma unroll
  for (int r = 0; r < 8; ++r) { m_run[r] = NEGMIN; l_run[r] = 0.f; }
  v8f acc[4] = {};

  const int nl = lane & 15;
  const int mb = (lane >> 4) << 3;
  const unsigned* mrow = mw + (size_t)(b * NSEQ) * NWORDS;

  for (int s0 = 0; s0 < NSEQ; s0 += 32) {
    __syncthreads();  // previous tile fully consumed before overwrite
    // --- cooperative stage: K 32x64 and V^T 64x32 (f16) into LDS ---
#pragma unroll
    for (int i = 0; i < 2; ++i) {
      const int idx = tid + i * 128;           // 256 x 16B segments
      const int row = idx >> 3, seg = idx & 7;
      *(v8h*)(ktile + row * 64 + seg * 8) = *(const v8h*)(
          kh + ((size_t)(b * NSEQ + s0 + row)) * KVE + h * DH + seg * 8);
    }
#pragma unroll
    for (int i = 0; i < 2; ++i) {
      const int idx = tid + i * 128;
      const int row = idx >> 2, seg = idx & 3;
      *(v8h*)(vtile + row * 32 + seg * 8) = *(const v8h*)(
          vth + ((size_t)(b * KVE + h * DH + row)) * NSEQ + s0 + seg * 8);
    }
    __syncthreads();  // staged tile visible to all waves

    // --- sim tile: two 16-wide QK^T tiles over D=64 (2 WMMAs each) ---
    v8f sim[2];
#pragma unroll
    for (int u = 0; u < 2; ++u) {
      v16h kb0 = load_b_frag(ktile + u * 16 * 64, 64, lane);
      v16h kb1 = load_b_frag(ktile + u * 16 * 64 + 32, 64, lane);
      v8f c = {};
      c = wmma_f16(qa0, kb0, c);
      c = wmma_f16(qa1, kb1, c);
      sim[u] = c;
    }

    // --- packed mask words: one u32 covers both 16-col halves of a row ---
    unsigned wb[8];
#pragma unroll
    for (int r = 0; r < 8; ++r)
      wb[r] = mrow[(size_t)(n0 + mb + r) * NWORDS + (s0 >> 5)];

    // --- online softmax, write P tile to this wave's LDS slice ---
#pragma unroll
    for (int r = 0; r < 8; ++r) {
      const bool c0 = (wb[r] >> nl) & 1u;
      const bool c1 = (wb[r] >> (nl + 16)) & 1u;
      const float v0 = c0 ? sim[0][r] : NEGMIN;
      const float v1 = c1 ? sim[1][r] : NEGMIN;
      const float tmax = half_reduce_max(fmaxf(v0, v1));
      const float m_new = fmaxf(m_run[r], tmax);
      const float rescale = __expf(m_run[r] - m_new);
      const float p0 = __expf(v0 - m_new);
      const float p1 = __expf(v1 - m_new);
      l_run[r] = l_run[r] * rescale + half_reduce_sum(p0 + p1);
      m_run[r] = m_new;
#pragma unroll
      for (int t4 = 0; t4 < 4; ++t4) acc[t4][r] *= rescale;
      pt[(mb + r) * 32 + nl]      = (h16)p0;   // 16x32 row-major P tile
      pt[(mb + r) * 32 + 16 + nl] = (h16)p1;
    }

    // --- P(16x32) @ V(32x64): reload P as A-frag (LDS in-order per wave) ---
    v16h pa = load_a_frag(pt, 32, lane);
#pragma unroll
    for (int t4 = 0; t4 < 4; ++t4) {
      v16h bf = load_b_frag(vtile + t4 * 16 * 32, 32, lane);
      acc[t4] = wmma_f16(pa, bf, acc[t4]);
    }
  }

  // --- normalize and store (fp32) ---
  float invl[8];
#pragma unroll
  for (int r = 0; r < 8; ++r) invl[r] = 1.0f / l_run[r];
#pragma unroll
  for (int t4 = 0; t4 < 4; ++t4) {
#pragma unroll
    for (int r = 0; r < 8; ++r) {
      const int n = n0 + mb + r;
      attn_out[((size_t)(b * NSEQ + n)) * EDIM + hq * DH + t4 * 16 + nl] =
          acc[t4][r] * invl[r];
    }
  }
}

// ---------------------------------------------------------------------------
// LayerNorm over E=1024, one block per token row; fp16 output for final GEMM.
// ---------------------------------------------------------------------------
__global__ __launch_bounds__(256) void ln_kernel(const float* __restrict__ x,
                                                 const float* __restrict__ g,
                                                 const float* __restrict__ bta,
                                                 h16* __restrict__ y) {
  const int row = blockIdx.x;
  const int tid = threadIdx.x;
  const float* xr = x + (size_t)row * EDIM;
  __shared__ float red[256];

  float v[4], s = 0.f;
#pragma unroll
  for (int i = 0; i < 4; ++i) { v[i] = xr[tid + i * 256]; s += v[i]; }
  red[tid] = s;
  __syncthreads();
  for (int off = 128; off; off >>= 1) {
    if (tid < off) red[tid] += red[tid + off];
    __syncthreads();
  }
  const float mu = red[0] * (1.f / EDIM);
  __syncthreads();

  float s2 = 0.f;
#pragma unroll
  for (int i = 0; i < 4; ++i) { const float d = v[i] - mu; s2 += d * d; }
  red[tid] = s2;
  __syncthreads();
  for (int off = 128; off; off >>= 1) {
    if (tid < off) red[tid] += red[tid + off];
    __syncthreads();
  }
  const float inv = rsqrtf(red[0] * (1.f / EDIM) + 1e-5f);
#pragma unroll
  for (int i = 0; i < 4; ++i) {
    const int c = tid + i * 256;
    y[(size_t)row * EDIM + c] = (h16)((v[i] - mu) * inv * g[c] + bta[c]);
  }
}

// ---------------------------------------------------------------------------
extern "C" void kernel_launch(void* const* d_in, const int* in_sizes, int n_in,
                              void* d_out, int out_size, void* d_ws,
                              size_t ws_size, hipStream_t stream) {
  const float* query = (const float*)d_in[0];
  const float* keyi  = (const float*)d_in[1];
  const float* value = (const float*)d_in[2];
  const int*   adj   = (const int*)d_in[3];
  const float* Wq = (const float*)d_in[4];
  const float* bq = (const float*)d_in[5];
  const float* Wk = (const float*)d_in[6];
  const float* bk = (const float*)d_in[7];
  const float* Wv = (const float*)d_in[8];
  const float* bv = (const float*)d_in[9];
  const float* Wo = (const float*)d_in[10];
  const float* bo = (const float*)d_in[11];
  const float* lng = (const float*)d_in[12];
  const float* lnb = (const float*)d_in[13];
  float* out = (float*)d_out;

  char* ws = (char*)d_ws;
  const size_t MB = 1024ull * 1024ull;
  h16*      xq  = (h16*)(ws + 0);                    // 8 MB (dead after Q proj)
  h16*      xk  = (h16*)(ws + 8 * MB);               // 8 MB (dead after K proj)
  h16*      xv  = (h16*)(ws + 16 * MB);              // 8 MB (dead after V proj)
  h16*      qh  = (h16*)(ws + 24 * MB);              // 8 MB
  h16*      kh  = (h16*)(ws + 32 * MB);              // 2 MB
  h16*      vth = (h16*)(ws + 34 * MB);              // 2 MB
  h16*      wqh = (h16*)(ws + 36 * MB);              // 2 MB
  h16*      wkh = (h16*)(ws + 38 * MB);              // 0.5 MB
  h16*      wvh = (h16*)(ws + 38 * MB + 512 * 1024); // 0.5 MB
  h16*      woh = (h16*)(ws + 39 * MB);              // 2 MB
  unsigned* mwp = (unsigned*)(ws + 41 * MB);         // 1 MB -> total 42 MB
  float*    attn = (float*)(ws + 0);                 // 16 MB, reuses xq+xk
  h16*      xo   = (h16*)(ws + 16 * MB);             // 8 MB, reuses xv

  const int NIN = MROWS * EDIM;  // 4194304
  cvt_f32_f16<<<(NIN + 255) / 256, 256, 0, stream>>>(query, xq, NIN);
  cvt_f32_f16<<<(NIN + 255) / 256, 256, 0, stream>>>(keyi, xk, NIN);
  cvt_f32_f16<<<(NIN + 255) / 256, 256, 0, stream>>>(value, xv, NIN);
  cvt_f32_f16<<<(EDIM * EDIM + 255) / 256, 256, 0, stream>>>(Wq, wqh, EDIM * EDIM);
  cvt_f32_f16<<<(KVE * EDIM + 255) / 256, 256, 0, stream>>>(Wk, wkh, KVE * EDIM);
  cvt_f32_f16<<<(KVE * EDIM + 255) / 256, 256, 0, stream>>>(Wv, wvh, KVE * EDIM);
  cvt_f32_f16<<<(EDIM * EDIM + 255) / 256, 256, 0, stream>>>(Wo, woh, EDIM * EDIM);
  pack_mask<<<BB * NSEQ * NWORDS / 256, 256, 0, stream>>>(adj, mwp);

  // Q: scale by 1/sqrt(64) folded in (bias scaled too, matching reference)
  gemm_wmma<0><<<(MROWS / 16) * (EDIM / 64) / 8, 256, 0, stream>>>(
      xq, wqh, bq, qh, MROWS, EDIM, EDIM, 0.125f);
  gemm_wmma<0><<<(MROWS / 16) * (KVE / 64) / 8, 256, 0, stream>>>(
      xk, wkh, bk, kh, MROWS, EDIM, KVE, 1.0f);
  gemm_wmma<1><<<(MROWS / 16) * (KVE / 64) / 8, 256, 0, stream>>>(
      xv, wvh, bv, vth, MROWS, EDIM, KVE, 1.0f);

  // attention: B*HQ*(N/16) = 4096 waves, 4 waves (1 block) per (b,hq,4 tiles)
  attn_wmma<<<BB * HQn * (NSEQ / 16) / 4, 128, 0, stream>>>(qh, kh, vth, mwp,
                                                            attn);
  ln_kernel<<<MROWS, 256, 0, stream>>>(attn, lng, lnb, xo);

  gemm_wmma<2><<<(MROWS / 16) * (EDIM / 64) / 8, 256, 0, stream>>>(
      xo, woh, bo, out, MROWS, EDIM, EDIM, 1.0f);
}
